// SpatialGATLargeWithJointEncoder_17016660427302
// MI455X (gfx1250) — matrix operand order repaired and checked by reading, not compiled, not run on hardware
//
#include <hip/hip_runtime.h>

// ---------------------------------------------------------------------------
// SpatialGATLargeWithJointEncoder forward for MI455X (gfx1250, wave32, WMMA)
//
//  * Dense layers -> bf16 WMMA GEMM (v_wmma_f32_16x16x32_bf16), f32 accum,
//    fused BN+bias(+ReLU) epilogue writing bf16 for the next GEMM.
//  * GEMM staging uses CDNA5 async global->LDS copies (ASYNCcnt) with
//    double-buffered LDS so tile t+1 copies overlap tile t WMMAs.
//  * Edge path (the real ~3 GB bandwidth bottleneck) uses float4 gathers and
//    f32 L2 scatter-atomics; segment-max via order-preserving u32 atomicMax.
// ---------------------------------------------------------------------------

typedef __attribute__((ext_vector_type(16))) __bf16 bf16x16;
typedef __attribute__((ext_vector_type(8)))  float  f32x8;
typedef __attribute__((ext_vector_type(4)))  int    i32x4;

#define NN_NODES 40000
#define NE_EDGES 320000

// ---------------- small helpers ----------------
__device__ __forceinline__ unsigned short f2bf(float f) {
  unsigned u = __float_as_uint(f);
  u += 0x7fffu + ((u >> 16) & 1u);          // round-to-nearest-even
  return (unsigned short)(u >> 16);
}
__device__ __forceinline__ float bf2f(unsigned short h) {
  return __uint_as_float((unsigned)h << 16);
}
__device__ __forceinline__ unsigned encOrd(float x) {
  unsigned u = __float_as_uint(x);
  return (u & 0x80000000u) ? ~u : (u | 0x80000000u);
}
__device__ __forceinline__ float decOrd(unsigned e) {
  return __uint_as_float((e & 0x80000000u) ? (e & 0x7fffffffu) : ~e);
}

// ---------------- CDNA5 async global->LDS copy (16 bytes) ----------------
__device__ __forceinline__ void lds_async_b128(const unsigned short* g, unsigned short* l) {
#if __has_builtin(__builtin_amdgcn_global_load_async_to_lds_b128)
  __builtin_amdgcn_global_load_async_to_lds_b128(
      (__attribute__((address_space(1))) i32x4*)(void*)g,
      (__attribute__((address_space(3))) i32x4*)(void*)l, 0, 0);
#else
  // GV form: vdst = LDS byte offset, vaddr = 64-bit global address, saddr = off
  unsigned loff = (unsigned)(unsigned long long)(__attribute__((address_space(3))) unsigned short*)l;
  unsigned long long ga = (unsigned long long)g;
  asm volatile("global_load_async_to_lds_b128 %0, %1, off"
               :: "v"(loff), "v"(ga) : "memory");
#endif
}
__device__ __forceinline__ void wait_async0() {
#if __has_builtin(__builtin_amdgcn_s_wait_asynccnt)
  __builtin_amdgcn_s_wait_asynccnt(0);
#else
  asm volatile("s_wait_asynccnt 0x0" ::: "memory");
#endif
}

// ---------------- f32 -> bf16 convert with zero padding ----------------
__global__ void cvt_pad_bf16(const float* __restrict__ src,
                             unsigned short* __restrict__ dst,
                             int rowsDst, int colsDst, int rowsSrc, int colsSrc,
                             int srcStride, int srcColOff) {
  long long idx = (long long)blockIdx.x * blockDim.x + threadIdx.x;
  long long total = (long long)rowsDst * colsDst;
  if (idx >= total) return;
  int r = (int)(idx / colsDst), c = (int)(idx % colsDst);
  float v = 0.f;
  if (r < rowsSrc && c < colsSrc) v = src[(size_t)r * srcStride + srcColOff + c];
  dst[idx] = f2bf(v);
}

// ---------------- fold BN (+bias) into per-column scale/shift ----------------
__global__ void prep_scaleshift(const float* __restrict__ g, const float* __restrict__ be,
                                const float* __restrict__ m, const float* __restrict__ v,
                                const float* __restrict__ b,
                                float* __restrict__ scale, float* __restrict__ shift, int n) {
  int i = blockIdx.x * blockDim.x + threadIdx.x;
  if (i >= n) return;
  float s = 1.f, sh;
  float bias = b ? b[i] : 0.f;
  if (g) {
    s  = g[i] * rsqrtf(v[i] + 1e-5f);
    sh = (bias - m[i]) * s + be[i];
  } else {
    sh = bias;
  }
  scale[i] = s;
  shift[i] = sh;
}

// ---------------- generic u32 fill ----------------
__global__ void fill_u32(unsigned* __restrict__ p, unsigned val, long long n) {
  long long i = (long long)blockIdx.x * blockDim.x + threadIdx.x;
  if (i < n) p[i] = val;
}

// ---------------- bf16 WMMA GEMM (double-buffered, async staging) ----------------
// C[M,N] = A[M,K](bf16) @ B[K,N](bf16); K multiple of 32, N multiple of 64.
// Epilogue: y = acc*scale[n] + shift[n] (if scale), optional ReLU,
// written as f32 (outF) or bf16 (outB) at row stride ldc, column offset colOff.
#define BM 128
#define BN 64
#define BK 32
#define LDSP 48   // padded LDS row pitch (ushorts): 96 B, 16B-aligned, spreads banks

__global__ __launch_bounds__(256)
void gemm_bf16_wmma(const unsigned short* __restrict__ A,
                    const unsigned short* __restrict__ B,
                    float* __restrict__ outF, unsigned short* __restrict__ outB,
                    const float* __restrict__ scale, const float* __restrict__ shift,
                    int M, int N, int K, int ldc, int colOff, int relu) {
  __shared__ unsigned short As[2][BM * LDSP];  // 2 x 12 KB
  __shared__ unsigned short Bs[2][BN * LDSP];  // 2 x 6 KB (transposed: [n][k])

  const int tid   = threadIdx.x;
  const int lane  = tid & 31;
  const int wave  = tid >> 5;       // 8 waves
  const int wm    = wave & 3;       // 4 waves along M -> 128
  const int wn    = wave >> 2;      // 2 waves along N -> 64
  const int m0    = blockIdx.y * BM;
  const int n0    = blockIdx.x * BN;
  const int lhalf = lane >> 4;
  const int l16   = lane & 15;

  // per-thread staging coordinates
  const int ar0 = (tid * 2) >> 2;            // A chunk 0 row
  const int ac0 = ((tid * 2) & 3) * 8;       // A chunk 0 col
  const int ar1 = (tid * 2 + 1) >> 2;
  const int ac1 = ((tid * 2 + 1) & 3) * 8;
  const int bkr = tid >> 3;                  // B: k row 0..31
  const int bnc = (tid & 7) * 8;             // B: n chunk

  f32x8 acc[2][2];
#pragma unroll
  for (int i = 0; i < 2; ++i)
#pragma unroll
    for (int j = 0; j < 2; ++j) acc[i][j] = (f32x8)0.0f;

  const int T = K / BK;

  // --- prologue: stage tile 0 into buffer 0 ---
  {
    uint4 bv = *(const uint4*)(B + (size_t)bkr * N + n0 + bnc);
    int gm0 = m0 + ar0; if (gm0 >= M) gm0 = M - 1;   // clamp: padded rows discarded
    int gm1 = m0 + ar1; if (gm1 >= M) gm1 = M - 1;
    lds_async_b128(A + (size_t)gm0 * K + ac0, &As[0][ar0 * LDSP + ac0]);
    lds_async_b128(A + (size_t)gm1 * K + ac1, &As[0][ar1 * LDSP + ac1]);
    const unsigned short* pv = (const unsigned short*)&bv;
#pragma unroll
    for (int j = 0; j < 8; ++j) Bs[0][(bnc + j) * LDSP + bkr] = pv[j];
  }
  wait_async0();
  __syncthreads();

  for (int t = 0; t < T; ++t) {
    const int cur = t & 1, nxt = cur ^ 1;
    const bool more = (t + 1 < T);
    uint4 bv = make_uint4(0u, 0u, 0u, 0u);
    if (more) {
      const int k1 = (t + 1) * BK;
      bv = *(const uint4*)(B + (size_t)(k1 + bkr) * N + n0 + bnc);  // B -> regs (no wait yet)
      int gm0 = m0 + ar0; if (gm0 >= M) gm0 = M - 1;
      int gm1 = m0 + ar1; if (gm1 >= M) gm1 = M - 1;
      lds_async_b128(A + (size_t)gm0 * K + k1 + ac0, &As[nxt][ar0 * LDSP + ac0]);
      lds_async_b128(A + (size_t)gm1 * K + k1 + ac1, &As[nxt][ar1 * LDSP + ac1]);
    }

    // --- compute tile t from buffer cur (overlaps async copies above) ---
    // A 16x32 bf16 lane layout: lanes0-15: K0-7 & 16-23; lanes16-31: K8-15 & 24-31
#pragma unroll
    for (int i = 0; i < 2; ++i) {
      bf16x16 afrag;
      int arow = wm * 32 + i * 16 + l16;
      int kb   = lhalf * 8;
      ((uint4*)&afrag)[0] = *(const uint4*)(&As[cur][arow * LDSP + kb]);
      ((uint4*)&afrag)[1] = *(const uint4*)(&As[cur][arow * LDSP + kb + 16]);
#pragma unroll
      for (int j = 0; j < 2; ++j) {
        bf16x16 bfrag;
        int brow = wn * 32 + j * 16 + l16;
        int kbb  = lhalf * 8;
        ((uint4*)&bfrag)[0] = *(const uint4*)(&Bs[cur][brow * LDSP + kbb]);
        ((uint4*)&bfrag)[1] = *(const uint4*)(&Bs[cur][brow * LDSP + kbb + 16]);
        acc[i][j] = __builtin_amdgcn_wmma_f32_16x16x32_bf16(
            false, afrag, false, bfrag, (short)0, acc[i][j], false, false);
      }
    }

    if (more) {  // transpose-store next B tile after compute (loads have landed)
      const unsigned short* pv = (const unsigned short*)&bv;
#pragma unroll
      for (int j = 0; j < 8; ++j) Bs[nxt][(bnc + j) * LDSP + bkr] = pv[j];
    }
    wait_async0();
    __syncthreads();
  }

  // --- epilogue: D layout: lanes0-15 vgpr r -> (M=r,N=lane); lanes16-31 -> (M=r+8,N=lane-16)
#pragma unroll
  for (int i = 0; i < 2; ++i)
#pragma unroll
    for (int j = 0; j < 2; ++j)
#pragma unroll
      for (int r = 0; r < 8; ++r) {
        int mm = m0 + wm * 32 + i * 16 + r + lhalf * 8;
        int nn = n0 + wn * 32 + j * 16 + l16;
        if (mm < M && nn < N) {
          float vv = acc[i][j][r];
          if (scale) vv = vv * scale[nn] + shift[nn];
          if (relu && vv < 0.f) vv = 0.f;
          size_t o = (size_t)mm * ldc + colOff + nn;
          if (outB) outB[o] = f2bf(vv);
          else      outF[o] = vv;
        }
      }
}

// ---------------- attention coefficients: a_s/a_d per (node, head) ----------------
__global__ void attn_coef(const float* __restrict__ hW, const float* __restrict__ a_src,
                          const float* __restrict__ a_dst, float* __restrict__ asO,
                          float* __restrict__ adO, int Nn, int H, int C) {
  int idx = blockIdx.x * blockDim.x + threadIdx.x;
  if (idx >= Nn * H) return;
  int n = idx / H, h = idx % H;
  const float* row = hW + (size_t)n * H * C + (size_t)h * C;
  const float* av  = a_src + h * C;
  const float* dv  = a_dst + h * C;
  float s = 0.f, d = 0.f;
  for (int c = 0; c < C; ++c) { float x = row[c]; s += x * av[c]; d += x * dv[c]; }
  asO[idx] = s;
  adO[idx] = d;
}

// ---------------- edge pass 1: leaky-relu logits + segment max ----------------
__global__ void edge_logits(const int* __restrict__ ei, const float* __restrict__ asB,
                            const float* __restrict__ adB, float* __restrict__ ebuf,
                            unsigned* __restrict__ emax, int NE, int Nn, int H) {
  int idx = blockIdx.x * blockDim.x + threadIdx.x;
  int total = (NE + Nn) * H;
  if (idx >= total) return;
  int e = idx / H, h = idx % H;
  int s, d;
  if (e < NE) { s = ei[e]; d = ei[NE + e]; } else { s = d = e - NE; }
  float x = asB[s * H + h] + adB[d * H + h];
  x = (x > 0.f) ? x : 0.2f * x;
  ebuf[idx] = x;
  atomicMax(&emax[d * H + h], encOrd(x));
}

// ---------------- edge pass 2: exp(e - max) + segment sum ----------------
__global__ void edge_softmax_num(const int* __restrict__ ei, float* __restrict__ ebuf,
                                 const unsigned* __restrict__ emax, float* __restrict__ denom,
                                 int NE, int Nn, int H) {
  int idx = blockIdx.x * blockDim.x + threadIdx.x;
  int total = (NE + Nn) * H;
  if (idx >= total) return;
  int e = idx / H, h = idx % H;
  int d = (e < NE) ? ei[NE + e] : (e - NE);
  float ee = __expf(ebuf[idx] - decOrd(emax[d * H + h]));
  ebuf[idx] = ee;
  atomicAdd(&denom[d * H + h], ee);
}

// ---------------- edge pass 3: weighted message scatter-add (float4) ----------------
__global__ void edge_aggregate(const int* __restrict__ ei, const float* __restrict__ ebuf,
                               const float* __restrict__ denom, const float* __restrict__ hW,
                               float* __restrict__ agg, int NE, int Nn, int H, int C) {
  int HC = H * C;
  int fpv = HC >> 2;
  long long idx = (long long)blockIdx.x * blockDim.x + threadIdx.x;
  long long total = (long long)(NE + Nn) * fpv;
  if (idx >= total) return;
  int e  = (int)(idx / fpv);
  int f  = (int)(idx % fpv) * 4;
  int h  = f / C;
  int s, d;
  if (e < NE) { s = ei[e]; d = ei[NE + e]; } else { s = d = e - NE; }
  float alpha = ebuf[(size_t)e * H + h] / denom[d * H + h];
  const float4 m = *(const float4*)(hW + (size_t)s * HC + f);
  float* o = agg + (size_t)d * HC + f;
  atomicAdd(o + 0, m.x * alpha);
  atomicAdd(o + 1, m.y * alpha);
  atomicAdd(o + 2, m.z * alpha);
  atomicAdd(o + 3, m.w * alpha);
}

// ---------------- finalize GAT layer: +bias, ELU, store bf16 ----------------
__global__ void gat_finalize(const float* __restrict__ agg, const float* __restrict__ bias,
                             unsigned short* __restrict__ actB, int Nn, int D) {
  long long idx = (long long)blockIdx.x * blockDim.x + threadIdx.x;
  long long total = (long long)Nn * D;
  if (idx >= total) return;
  float v = agg[idx] + bias[idx % D];
  v = (v > 0.f) ? v : expm1f(v);
  actB[idx] = f2bf(v);
}

// ---------------- tiny final linear 64 -> 2 ----------------
__global__ void mlp2_kernel(const unsigned short* __restrict__ act, const float* __restrict__ W,
                            const float* __restrict__ b, float* __restrict__ out, int Nn) {
  int n = blockIdx.x * blockDim.x + threadIdx.x;
  if (n >= Nn) return;
  float o0 = b[0], o1 = b[1];
  const unsigned short* row = act + (size_t)n * 64;
  for (int k = 0; k < 64; ++k) {
    float a = bf2f(row[k]);
    o0 += a * W[k * 2 + 0];
    o1 += a * W[k * 2 + 1];
  }
  out[n * 2 + 0] = o0;
  out[n * 2 + 1] = o1;
}

// ===========================================================================
extern "C" void kernel_launch(void* const* d_in, const int* in_sizes, int n_in,
                              void* d_out, int out_size, void* d_ws, size_t ws_size,
                              hipStream_t stream) {
  (void)in_sizes; (void)n_in; (void)out_size; (void)ws_size;
  const int NN = NN_NODES, NE = NE_EDGES;

  // ----- inputs (setup_inputs dict order, nested insertion order) -----
  const float* x  = (const float*)d_in[0];
  const int*   ei = (const int*)d_in[1];
  int pi = 2;
  auto F = [&]() { return (const float*)d_in[pi++]; };
  const float *rna1W=F(), *rna1b=F();
  const float *bn1g=F(), *bn1be=F(), *bn1m=F(), *bn1v=F();
  const float *rna2W=F(), *rna2b=F();
  const float *bn2g=F(), *bn2be=F(), *bn2m=F(), *bn2v=F();
  const float *pro1W=F(), *pro1b=F();
  const float *pbn1g=F(), *pbn1be=F(), *pbn1m=F(), *pbn1v=F();
  const float *pro2W=F(), *pro2b=F();
  const float *pbn2g=F(), *pbn2be=F(), *pbn2m=F(), *pbn2v=F();
  const float *fusW=F(), *fusb=F();
  const float *fbng=F(), *fbnbe=F(), *fbnm=F(), *fbnv=F();
  const float *g1W=F(), *g1as=F(), *g1ad=F(), *g1b=F();
  const float *g2W=F(), *g2as=F(), *g2ad=F(), *g2b=F();
  const float *g3W=F(), *g3as=F(), *g3ad=F(), *g3b=F();
  const float *m1W=F(), *m1b=F();
  const float *m2W=F(), *m2b=F();

  // ----- workspace bump allocator -----
  size_t off = 0;
  auto alloc = [&](size_t bytes) -> void* {
    off = (off + 255) & ~(size_t)255;
    void* p = (char*)d_ws + off;
    off += bytes;
    return p;
  };
  typedef unsigned short us;
  us* rnaA  = (us*)alloc((size_t)NN * 416 * 2);
  us* proA  = (us*)alloc((size_t)NN * 32 * 2);
  us* wbuf  = (us*)alloc((size_t)262144 * 2);          // reused bf16 weight buffer
  float* scale = (float*)alloc(1024 * 4);
  float* shift = (float*)alloc(1024 * 4);
  us* r1    = (us*)alloc((size_t)NN * 256 * 2);
  us* p1    = (us*)alloc((size_t)NN * 128 * 2);
  us* hcat  = (us*)alloc((size_t)NN * 384 * 2);
  us* hfus  = (us*)alloc((size_t)NN * 256 * 2);
  us* g1act = (us*)alloc((size_t)NN * 1024 * 2);
  us* g2act = (us*)alloc((size_t)NN * 192 * 2);
  us* g3act = (us*)alloc((size_t)NN * 128 * 2);
  us* m1act = (us*)alloc((size_t)NN * 64 * 2);
  float* hW   = (float*)alloc((size_t)NN * 1024 * 4);  // reused across GAT layers
  float* agg  = (float*)alloc((size_t)NN * 1024 * 4);  // reused
  float* asB  = (float*)alloc((size_t)NN * 4 * 4);
  float* adB  = (float*)alloc((size_t)NN * 4 * 4);
  float* ebuf = (float*)alloc((size_t)(NE + NN) * 4 * 4);
  unsigned* emax  = (unsigned*)alloc((size_t)NN * 4 * 4);
  float*    denom = (float*)alloc((size_t)NN * 4 * 4);

  auto blk = [](long long n) { return (unsigned)((n + 255) / 256); };

  auto cvt = [&](const float* s, us* dgt, int rd, int cd, int rs, int cs, int stride, int coff) {
    cvt_pad_bf16<<<blk((long long)rd * cd), 256, 0, stream>>>(s, dgt, rd, cd, rs, cs, stride, coff);
  };
  auto prep = [&](const float* g, const float* be, const float* m, const float* v,
                  const float* b, int n) {
    prep_scaleshift<<<blk(n), 256, 0, stream>>>(g, be, m, v, b, scale, shift, n);
  };
  auto gemm = [&](const us* A, const us* B, float* oF, us* oB, const float* sc,
                  const float* sh, int M, int N, int K, int ldc, int coff, int relu) {
    dim3 g(N / 64, (M + 127) / 128);
    gemm_bf16_wmma<<<g, 256, 0, stream>>>(A, B, oF, oB, sc, sh, M, N, K, ldc, coff, relu);
  };

  // ----- encoder -----
  cvt(x, rnaA, NN, 416, NN, 405, 432, 0);             // rna features, K-pad 405->416
  cvt(x, proA, NN, 32, NN, 27, 432, 405);             // protein features, 27->32
  cvt(rna1W, wbuf, 416, 256, 405, 256, 256, 0);
  prep(bn1g, bn1be, bn1m, bn1v, rna1b, 256);
  gemm(rnaA, wbuf, nullptr, r1, scale, shift, NN, 256, 416, 256, 0, 1);
  cvt(rna2W, wbuf, 256, 256, 256, 256, 256, 0);
  prep(bn2g, bn2be, bn2m, bn2v, rna2b, 256);
  gemm(r1, wbuf, nullptr, hcat, scale, shift, NN, 256, 256, 384, 0, 1);     // h[:, :256]
  cvt(pro1W, wbuf, 32, 128, 27, 128, 128, 0);
  prep(pbn1g, pbn1be, pbn1m, pbn1v, pro1b, 128);
  gemm(proA, wbuf, nullptr, p1, scale, shift, NN, 128, 32, 128, 0, 1);
  cvt(pro2W, wbuf, 128, 128, 128, 128, 128, 0);
  prep(pbn2g, pbn2be, pbn2m, pbn2v, pro2b, 128);
  gemm(p1, wbuf, nullptr, hcat, scale, shift, NN, 128, 128, 384, 256, 1);   // h[:, 256:]
  cvt(fusW, wbuf, 384, 256, 384, 256, 256, 0);
  prep(fbng, fbnbe, fbnm, fbnv, fusb, 256);
  gemm(hcat, wbuf, nullptr, hfus, scale, shift, NN, 256, 384, 256, 0, 1);

  // ----- GAT layers -----
  auto gat = [&](const us* actIn, int Kin, const float* Wf, const float* aS,
                 const float* aD, const float* bias, int H, int C, us* actOut) {
    int HC = H * C;
    cvt(Wf, wbuf, Kin, HC, Kin, HC, HC, 0);
    gemm(actIn, wbuf, hW, nullptr, nullptr, nullptr, NN, HC, Kin, HC, 0, 0);
    attn_coef<<<blk((long long)NN * H), 256, 0, stream>>>(hW, aS, aD, asB, adB, NN, H, C);
    fill_u32<<<blk((long long)NN * H), 256, 0, stream>>>(emax, 0x007FFFFFu, (long long)NN * H); // enc(-inf)
    fill_u32<<<blk((long long)NN * H), 256, 0, stream>>>((unsigned*)denom, 0u, (long long)NN * H);
    fill_u32<<<blk((long long)NN * HC), 256, 0, stream>>>((unsigned*)agg, 0u, (long long)NN * HC);
    long long etot = (long long)(NE + NN) * H;
    edge_logits<<<blk(etot), 256, 0, stream>>>(ei, asB, adB, ebuf, emax, NE, NN, H);
    edge_softmax_num<<<blk(etot), 256, 0, stream>>>(ei, ebuf, emax, denom, NE, NN, H);
    long long atot = (long long)(NE + NN) * (HC / 4);
    edge_aggregate<<<blk(atot), 256, 0, stream>>>(ei, ebuf, denom, hW, agg, NE, NN, H, C);
    gat_finalize<<<blk((long long)NN * HC), 256, 0, stream>>>(agg, bias, actOut, NN, HC);
  };
  gat(hfus, 256, g1W, g1as, g1ad, g1b, 4, 256, g1act);
  gat(g1act, 1024, g2W, g2as, g2ad, g2b, 2, 96, g2act);
  gat(g2act, 192, g3W, g3as, g3ad, g3b, 1, 128, g3act);   // heads=1: mean == identity

  // ----- MLP head -----
  cvt(m1W, wbuf, 128, 64, 128, 64, 64, 0);
  prep(nullptr, nullptr, nullptr, nullptr, m1b, 64);
  gemm(g3act, wbuf, nullptr, m1act, scale, shift, NN, 64, 128, 64, 0, 1);
  mlp2_kernel<<<blk(NN), 256, 0, stream>>>(m1act, m2W, m2b, (float*)d_out, NN);
}
